// MoEClassifier_154618823176
// MI455X (gfx1250) — matrix operand order repaired and checked by reading, not compiled, not run on hardware
//
#include <hip/hip_runtime.h>
#include <hip/hip_bf16.h>

// ---------------------------------------------------------------------------
// MoE classifier forward for MI455X (gfx1250, wave32, WMMA bf16 16x16x32).
// Pipeline: bf16 pre-convert -> proj -> router(softmax/top2)
//           -> gated expert MLP (M=32 tile, 128KB LDS hid) -> classifier.
// ---------------------------------------------------------------------------

typedef __bf16 bf16_t;
typedef __attribute__((ext_vector_type(16))) __bf16 v16bf;
typedef __attribute__((ext_vector_type(8)))  __bf16 v8bf;
typedef __attribute__((ext_vector_type(8)))  float  v8f;

static constexpr int B_ = 8192;
static constexpr int F_ = 1024;
static constexpr int H_ = 1024;
static constexpr int E_ = 8;
static constexpr int D_ = 2048;
static constexpr int C_ = 1000;

#define DEVFN static __device__ __forceinline__

DEVFN v16bf cat8(v8bf lo, v8bf hi) {
  return __builtin_shufflevector(lo, hi, 0, 1, 2, 3, 4, 5, 6, 7,
                                         8, 9, 10, 11, 12, 13, 14, 15);
}
DEVFN v8bf cvt8(v8f f) { return __builtin_convertvector(f, v8bf); }

// A fragment (16x32 bf16, MxK), row-major source: lane%16 = M row; lanes<16
// hold K [k0,k0+16); lanes>=16 hold K [k0+16,k0+32).
DEVFN v16bf ldA_bf16(const bf16_t* __restrict__ base, int ld, int k0) {
  const int lane = threadIdx.x & 31;
  const int m = lane & 15, hf = lane >> 4;
  const bf16_t* p = base + (size_t)m * ld + k0 + hf * 16;
  v8bf lo = *(const v8bf*)p;
  v8bf hi = *(const v8bf*)(p + 8);
  return cat8(lo, hi);
}
// B fragment (32x16 bf16, KxN) from row-major W[n][k] (B[k][n]=W[n][k]):
// lane%16 = N; lanes<16 hold K [k0,k0+16), lanes>=16 hold K [k0+16,k0+32).
DEVFN v16bf ldB_bf16(const bf16_t* __restrict__ base, int ld, int k0) {
  const int lane = threadIdx.x & 31;
  const int n = lane & 15, hf = lane >> 4;
  const bf16_t* p = base + (size_t)n * ld + k0 + hf * 16;
  v8bf lo = *(const v8bf*)p;
  v8bf hi = *(const v8bf*)(p + 8);
  return cat8(lo, hi);
}
DEVFN v16bf ldB_bf16_clamp(const bf16_t* __restrict__ mat, int ld, int n0,
                           int maxrow, int k0) {
  const int lane = threadIdx.x & 31;
  int n = n0 + (lane & 15);
  if (n > maxrow) n = maxrow;
  const int hf = lane >> 4;
  const bf16_t* p = mat + (size_t)n * ld + k0 + hf * 16;
  v8bf lo = *(const v8bf*)p;
  v8bf hi = *(const v8bf*)(p + 8);
  return cat8(lo, hi);
}

DEVFN v8f wmma_bf16(v16bf a, v16bf b, v8f c) {
  return __builtin_amdgcn_wmma_f32_16x16x32_bf16(
      false, a, false, b, (short)0, c, false, false);
}
DEVFN float relu(float v) { return v > 0.f ? v : 0.f; }

// ---------------------------------------------------------------------------
// Kernel 0: fp32 -> bf16 conversion (one-time, vectorized x8).
// ---------------------------------------------------------------------------
__global__ __launch_bounds__(256) void cvt_bf16_kernel(
    const float* __restrict__ src, bf16_t* __restrict__ dst, int n8) {
  const int i = blockIdx.x * blockDim.x + threadIdx.x;
  if (i < n8) {
    v8f f = *(const v8f*)(src + (size_t)i * 8);
    *(v8bf*)(dst + (size_t)i * 8) = cvt8(f);
  }
}

// ---------------------------------------------------------------------------
// Kernel 1: h = relu(x @ Wp^T + bp)  [B,H] bf16. grid (B/16, H/128), 8 waves.
// ---------------------------------------------------------------------------
__global__ __launch_bounds__(256) void proj_kernel(
    const bf16_t* __restrict__ xb, const bf16_t* __restrict__ Wpb,
    const float* __restrict__ bp, bf16_t* __restrict__ hout) {
  const int m0 = blockIdx.x * 16;
  const int wave = threadIdx.x >> 5;
  const int lane = threadIdx.x & 31;
  const int n0 = blockIdx.y * 128 + wave * 16;

  const bf16_t* aBase = xb + (size_t)m0 * F_;
  const bf16_t* bBase = Wpb + (size_t)n0 * F_;
  v8f acc = {};
  for (int k0 = 0; k0 < F_; k0 += 32) {
    __builtin_prefetch((const void*)(bBase + k0 + 256), 0, 1);
    v16bf a = ldA_bf16(aBase, F_, k0);
    v16bf b = ldB_bf16(bBase, F_, k0);
    acc = wmma_bf16(a, b, acc);
  }
  const int n = n0 + (lane & 15);
  const int hf = lane >> 4;
  const float bias = bp[n];
#pragma unroll
  for (int r = 0; r < 8; ++r)
    hout[(size_t)(m0 + r + hf * 8) * H_ + n] = (bf16_t)relu(acc[r] + bias);
}

// ---------------------------------------------------------------------------
// Kernel 2: router. Per-token logits over E=8, softmax, top-2 + renorm.
// ---------------------------------------------------------------------------
__global__ __launch_bounds__(256) void router_kernel(
    const bf16_t* __restrict__ h, const float* __restrict__ Wg,
    float* __restrict__ probs, float* __restrict__ gate_w,
    int* __restrict__ gate_i) {
  __shared__ float wg_s[E_ * H_];  // 32 KB
  for (int i = threadIdx.x; i < E_ * H_; i += blockDim.x) wg_s[i] = Wg[i];
  __syncthreads();

  const int b = blockIdx.x * blockDim.x + threadIdx.x;
  const bf16_t* hr = h + (size_t)b * H_;

  float acc[E_];
#pragma unroll
  for (int e = 0; e < E_; ++e) acc[e] = 0.f;
  for (int f = 0; f < H_; ++f) {
    const float hv = (float)hr[f];
#pragma unroll
    for (int e = 0; e < E_; ++e) acc[e] += hv * wg_s[e * H_ + f];
  }
  float mx = acc[0];
#pragma unroll
  for (int e = 1; e < E_; ++e) mx = fmaxf(mx, acc[e]);
  float p[E_], s = 0.f;
#pragma unroll
  for (int e = 0; e < E_; ++e) { p[e] = __expf(acc[e] - mx); s += p[e]; }
  const float inv = 1.f / s;
#pragma unroll
  for (int e = 0; e < E_; ++e) {
    p[e] *= inv;
    probs[(size_t)b * E_ + e] = p[e];
  }
  int i0 = 0; float p0 = p[0];
#pragma unroll
  for (int e = 1; e < E_; ++e) if (p[e] > p0) { p0 = p[e]; i0 = e; }
  int i1 = (i0 == 0) ? 1 : 0; float p1 = p[i1];
#pragma unroll
  for (int e = 0; e < E_; ++e)
    if (e != i0 && p[e] > p1) { p1 = p[e]; i1 = e; }
  const float wsum = p0 + p1;
  gate_w[2 * b + 0] = p0 / wsum;
  gate_w[2 * b + 1] = p1 / wsum;
  gate_i[2 * b + 0] = i0;
  gate_i[2 * b + 1] = i1;
}

// ---------------------------------------------------------------------------
// Kernel 3: gated expert MLP, M_TILE=32 tokens. grid B/32, 8 waves.
// phase A: hid = relu(h_tile @ W1[e]^T + b1) -> 32x2048 bf16 in LDS (128 KB)
// phase B: out += gate * (hid @ W2[e]^T + b2) -> registers (16 v8f/lane)
// Two A fragments share each B fragment (2 WMMAs per weight load).
// ---------------------------------------------------------------------------
__global__ __launch_bounds__(256) void expert_kernel(
    const bf16_t* __restrict__ h, const bf16_t* __restrict__ W1b,
    const float* __restrict__ b1, const bf16_t* __restrict__ W2b,
    const float* __restrict__ b2, const float* __restrict__ gate_w,
    const int* __restrict__ gate_i, bf16_t* __restrict__ mo) {
  __shared__ bf16_t hid_s[32 * D_];  // 128 KB

  const int m0 = blockIdx.x * 32;
  const int wave = threadIdx.x >> 5;
  const int lane = threadIdx.x & 31;
  const int hf = lane >> 4;
  const bf16_t* aBase = h + (size_t)m0 * H_;

  v8f outacc[8][2] = {};  // [n-tile][m-sub]: wave's 32x128 out slice, fp32

  for (int e = 0; e < E_; ++e) {
    // Per-lane gates for the rows this lane-half owns; tile-active test.
    float garr[2][8];
    int anyg = 0;
#pragma unroll
    for (int ms = 0; ms < 2; ++ms)
#pragma unroll
      for (int r = 0; r < 8; ++r) {
        const int tok = m0 + ms * 16 + r + hf * 8;
        float g = 0.f;
        if (gate_i[2 * tok + 0] == e) g += gate_w[2 * tok + 0];
        if (gate_i[2 * tok + 1] == e) g += gate_w[2 * tok + 1];
        garr[ms][r] = g;
        anyg |= (g != 0.f);
      }
    // Barrier (also orders prior phase-B hid reads vs. new phase-A writes).
    if (!__syncthreads_or(anyg)) continue;  // expert unused by this tile

    // --- phase A: hid[32, wave*256 .. +256) ---
    for (int nt = 0; nt < 16; ++nt) {
      const int n0 = wave * 256 + nt * 16;
      const bf16_t* bBase = W1b + ((size_t)e * D_ + n0) * H_;
      v8f acc0 = {}, acc1 = {};
      for (int k0 = 0; k0 < H_; k0 += 32) {
        __builtin_prefetch((const void*)(bBase + k0 + 256), 0, 1);
        v16bf b  = ldB_bf16(bBase, H_, k0);
        v16bf a0 = ldA_bf16(aBase, H_, k0);
        v16bf a1 = ldA_bf16(aBase + 16 * H_, H_, k0);
        acc0 = wmma_bf16(a0, b, acc0);
        acc1 = wmma_bf16(a1, b, acc1);
      }
      const int n = n0 + (lane & 15);
      const float bias = b1[(size_t)e * D_ + n];
#pragma unroll
      for (int r = 0; r < 8; ++r) {
        hid_s[(r + hf * 8) * D_ + n]      = (bf16_t)relu(acc0[r] + bias);
        hid_s[(16 + r + hf * 8) * D_ + n] = (bf16_t)relu(acc1[r] + bias);
      }
    }
    __syncthreads();

    // --- phase B: out[32, wave*128 .. +128) += gate * (hid @ W2^T + b2) ---
    for (int nt = 0; nt < 8; ++nt) {
      const int n0 = wave * 128 + nt * 16;
      const bf16_t* bBase = W2b + ((size_t)e * H_ + n0) * D_;
      v8f acc0 = {}, acc1 = {};
      for (int k0 = 0; k0 < D_; k0 += 32) {
        __builtin_prefetch((const void*)(bBase + k0 + 256), 0, 1);
        v16bf b  = ldB_bf16(bBase, D_, k0);
        v16bf a0 = ldA_bf16(hid_s, D_, k0);
        v16bf a1 = ldA_bf16(hid_s + 16 * D_, D_, k0);
        acc0 = wmma_bf16(a0, b, acc0);
        acc1 = wmma_bf16(a1, b, acc1);
      }
      const int n = n0 + (lane & 15);
      const float bias = b2[(size_t)e * H_ + n];
#pragma unroll
      for (int r = 0; r < 8; ++r) {
        outacc[nt][0][r] += garr[0][r] * (acc0[r] + bias);
        outacc[nt][1][r] += garr[1][r] * (acc1[r] + bias);
      }
    }
  }

  // relu + store bf16
  for (int nt = 0; nt < 8; ++nt) {
    const int n = wave * 128 + nt * 16 + (lane & 15);
#pragma unroll
    for (int ms = 0; ms < 2; ++ms)
#pragma unroll
      for (int r = 0; r < 8; ++r)
        mo[(size_t)(m0 + ms * 16 + r + hf * 8) * H_ + n] =
            (bf16_t)relu(outacc[nt][ms][r]);
  }
}

// ---------------------------------------------------------------------------
// Kernel 4: logits = moe_out @ Wc^T + bc  [B,C], C=1000 edge-masked.
// ---------------------------------------------------------------------------
__global__ __launch_bounds__(256) void cls_kernel(
    const bf16_t* __restrict__ mo, const bf16_t* __restrict__ Wcb,
    const float* __restrict__ bc, float* __restrict__ logits) {
  const int m0 = blockIdx.x * 16;
  const int wave = threadIdx.x >> 5;
  const int lane = threadIdx.x & 31;
  const int n0 = blockIdx.y * 128 + wave * 16;

  const bf16_t* aBase = mo + (size_t)m0 * H_;
  v8f acc = {};
  for (int k0 = 0; k0 < H_; k0 += 32) {
    __builtin_prefetch((const void*)(Wcb + (size_t)n0 * H_ + k0 + 256), 0, 1);
    v16bf a = ldA_bf16(aBase, H_, k0);
    v16bf b = ldB_bf16_clamp(Wcb, H_, n0, C_ - 1, k0);
    acc = wmma_bf16(a, b, acc);
  }
  const int n = n0 + (lane & 15);
  if (n < C_) {
    const int hf = lane >> 4;
    const float bias = bc[n];
#pragma unroll
    for (int r = 0; r < 8; ++r)
      logits[(size_t)(m0 + r + hf * 8) * C_ + n] = acc[r] + bias;
  }
}

// ---------------------------------------------------------------------------
extern "C" void kernel_launch(void* const* d_in, const int* in_sizes, int n_in,
                              void* d_out, int out_size, void* d_ws,
                              size_t ws_size, hipStream_t stream) {
  (void)in_sizes; (void)n_in; (void)out_size; (void)ws_size;
  const float* x  = (const float*)d_in[0];
  const float* Wp = (const float*)d_in[1];
  const float* bp = (const float*)d_in[2];
  const float* Wg = (const float*)d_in[3];
  const float* W1 = (const float*)d_in[4];
  const float* b1 = (const float*)d_in[5];
  const float* W2 = (const float*)d_in[6];
  const float* b2 = (const float*)d_in[7];
  const float* Wc = (const float*)d_in[8];
  const float* bc = (const float*)d_in[9];
  // d_in[10] = top_k == 2 (compile-time specialized)

  float* logits = (float*)d_out;             // [B, C]
  float* probs  = logits + (size_t)B_ * C_;  // [B, E]

  // Workspace layout (bf16 staging + gates), ~120 MB.
  char* ws = (char*)d_ws;
  size_t off = 0;
  bf16_t* h_bf  = (bf16_t*)(ws + off); off += (size_t)B_ * H_ * 2;
  bf16_t* x_bf  = (bf16_t*)(ws + off); off += (size_t)B_ * F_ * 2;
  bf16_t* mo_bf = (bf16_t*)(ws + off); off += (size_t)B_ * H_ * 2;
  bf16_t* Wp_bf = (bf16_t*)(ws + off); off += (size_t)H_ * F_ * 2;
  bf16_t* W1_bf = (bf16_t*)(ws + off); off += (size_t)E_ * D_ * H_ * 2;
  bf16_t* W2_bf = (bf16_t*)(ws + off); off += (size_t)E_ * H_ * D_ * 2;
  bf16_t* Wc_bf = (bf16_t*)(ws + off); off += (size_t)C_ * H_ * 2;
  float*  gate_w = (float*)(ws + off); off += (size_t)B_ * 2 * 4;
  int*    gate_i = (int*)(ws + off);   off += (size_t)B_ * 2 * 4;

  auto cvt = [&](const float* s, bf16_t* d, size_t n) {
    const int n8 = (int)(n / 8);
    cvt_bf16_kernel<<<(n8 + 255) / 256, 256, 0, stream>>>(s, d, n8);
  };
  cvt(x,  x_bf,  (size_t)B_ * F_);
  cvt(Wp, Wp_bf, (size_t)H_ * F_);
  cvt(W1, W1_bf, (size_t)E_ * D_ * H_);
  cvt(W2, W2_bf, (size_t)E_ * H_ * D_);
  cvt(Wc, Wc_bf, (size_t)C_ * H_);

  proj_kernel<<<dim3(B_ / 16, H_ / 128), 256, 0, stream>>>(x_bf, Wp_bf, bp,
                                                           h_bf);
  router_kernel<<<dim3(B_ / 256), 256, 0, stream>>>(h_bf, Wg, probs, gate_w,
                                                    gate_i);
  expert_kernel<<<dim3(B_ / 32), 256, 0, stream>>>(h_bf, W1_bf, b1, W2_bf, b2,
                                                   gate_w, gate_i, mo_bf);
  cls_kernel<<<dim3(B_ / 16, 8), 256, 0, stream>>>(mo_bf, Wc_bf, bc, logits);
}
